// TAGConvNet_14388140441686
// MI455X (gfx1250) — matrix operand order repaired and checked by reading, not compiled, not run on hardware
//
#include <hip/hip_runtime.h>
#include <hip/hip_bf16.h>

// ---------------------------------------------------------------------------
// TAGConvNet for MI455X (gfx1250, wave32, WMMA + TDM).
// Dense 128x128 GEMMs -> v_wmma_f32_16x16x32_bf16 (bf16 inputs, f32 accum),
// A-tile staged to LDS by the Tensor Data Mover (tensor_load_to_lds +
// s_wait_tensorcnt), weights pre-transposed to bf16 [N,K].
// Sparse propagation -> float4 gather + global_atomic_add_f32 scatter
// (h and t both fit in the 192MB L2).
// ---------------------------------------------------------------------------

typedef __attribute__((ext_vector_type(16))) __bf16        v16bf;
typedef __attribute__((ext_vector_type(8)))  float         v8f;
typedef __attribute__((ext_vector_type(4))) unsigned int   v4u;

union Frag16 {
    v16bf          v;
    v4u            q[2];
    unsigned short u[16];
};

__device__ __forceinline__ unsigned short f2bf(float f) {
    // round-to-nearest-even fp32 -> bf16
    unsigned int u = __float_as_uint(f);
    u += 0x7FFFu + ((u >> 16) & 1u);
    return (unsigned short)(u >> 16);
}

// --------------------------- TDM support -----------------------------------
#if defined(__has_builtin)
#if __has_builtin(__builtin_amdgcn_tensor_load_to_lds)
#define HAVE_TDM 1
#endif
#endif

#ifdef HAVE_TDM
typedef __attribute__((ext_vector_type(4))) unsigned int tdm_u4;
typedef __attribute__((ext_vector_type(8))) int          tdm_i8;
typedef __attribute__((ext_vector_type(4))) int          tdm_i4;

// 2D tile load: global (row-major, fp32) -> LDS. Dims/strides in elements.
// D# packing per CDNA5 ISA:
//   group0: [1:0]=count=1, [63:32]=lds_addr, [120:64]=global_addr, [127:126]=type=2
//   group1: [17:16]=data_size(2 -> 4B), [79:48]=tensor_dim0, [111:80]=tensor_dim1,
//           [127:112]=tile_dim0, [143:128]=tile_dim1, [207:160]=tensor_dim0_stride
__device__ __forceinline__ void tdm_load_tile_f32(
    const float* gptr, unsigned lds_off_bytes,
    unsigned tensor_d0, unsigned tensor_d1, unsigned stride_d0,
    unsigned tile_d0, unsigned tile_d1)
{
    unsigned long long ga = (unsigned long long)(uintptr_t)gptr;
    tdm_u4 g0;
    g0.x = 0x1u;                                   // count=1, user descriptor
    g0.y = lds_off_bytes;                          // lds_addr (bytes)
    g0.z = (unsigned)(ga & 0xFFFFFFFFu);           // global_addr[31:0]
    g0.w = (unsigned)((ga >> 32) & 0x01FFFFFFu)    // global_addr[56:32]
         | 0x80000000u;                            // type = 2 ("image")
    tdm_i8 g1;
    g1[0] = 0x00020000;                            // data_size=4B, wg_mask=0
    g1[1] = (int)((tensor_d0 & 0xFFFFu) << 16);    // tensor_dim0[15:0]
    g1[2] = (int)(((tensor_d0 >> 16) & 0xFFFFu) | ((tensor_d1 & 0xFFFFu) << 16));
    g1[3] = (int)(((tensor_d1 >> 16) & 0xFFFFu) | ((tile_d0 & 0xFFFFu) << 16));
    g1[4] = (int)(tile_d1 & 0xFFFFu);              // tile_dim1; tile_dim2=0
    g1[5] = (int)stride_d0;                        // tensor_dim0_stride[31:0]
    g1[6] = 0;
    g1[7] = 0;
    tdm_i4 gz = {0, 0, 0, 0};
#if __clang_major__ >= 23
    tdm_i8 gz8 = {0, 0, 0, 0, 0, 0, 0, 0};
    __builtin_amdgcn_tensor_load_to_lds(g0, g1, gz, gz, gz8, 0);
#else
    __builtin_amdgcn_tensor_load_to_lds(g0, g1, gz, gz, 0);
#endif
}

__device__ __forceinline__ void tdm_wait() {
#if __has_builtin(__builtin_amdgcn_s_wait_tensorcnt)
    __builtin_amdgcn_s_wait_tensorcnt(0);
#else
    asm volatile("s_wait_tensorcnt 0x0" ::: "memory");
#endif
}
#endif // HAVE_TDM

// ---------------------------------------------------------------------------
// Zero a float buffer (grid-stride)
// ---------------------------------------------------------------------------
__global__ void k_zero(float* __restrict__ p, long long n) {
    long long i = (long long)blockIdx.x * blockDim.x + threadIdx.x;
    long long stride = (long long)gridDim.x * blockDim.x;
    for (; i < n; i += stride) p[i] = 0.0f;
}

// ---------------------------------------------------------------------------
// Convert [K,N] fp32 weight -> [N,K] bf16 (transposed so B fragments are
// contiguous 16B loads per lane)
// ---------------------------------------------------------------------------
__global__ void k_convert_wt(const float* __restrict__ W, unsigned short* __restrict__ Wt,
                             int K, int N) {
    int idx = blockIdx.x * blockDim.x + threadIdx.x;
    if (idx >= K * N) return;
    int n = idx / K;
    int k = idx - n * K;
    Wt[idx] = f2bf(W[k * N + n]);
}

// ---------------------------------------------------------------------------
// Degree count (segment_sum of ones over dst) and symmetric norm
// ---------------------------------------------------------------------------
__global__ void k_deg(const int* __restrict__ dst, float* __restrict__ deg, int n_edges) {
    int e = blockIdx.x * blockDim.x + threadIdx.x;
    if (e < n_edges) atomicAdd(&deg[dst[e]], 1.0f);
}

__global__ void k_dinv(const float* __restrict__ deg, float* __restrict__ dinv, int n) {
    int i = blockIdx.x * blockDim.x + threadIdx.x;
    if (i < n) {
        float d = deg[i];
        dinv[i] = (d > 0.0f) ? rsqrtf(d) : 0.0f;
    }
}

// ---------------------------------------------------------------------------
// lin0: [N,16] @ [16,128] + b0, ReLU  (tiny K; plain VALU kernel)
// ---------------------------------------------------------------------------
__global__ void k_lin0(const float* __restrict__ x, const float* __restrict__ W0,
                       const float* __restrict__ b0, float* __restrict__ out, int n_nodes) {
    int idx = blockIdx.x * blockDim.x + threadIdx.x;
    if (idx >= n_nodes * 128) return;
    int node = idx >> 7;
    int f    = idx & 127;
    float acc = b0[f];
#pragma unroll
    for (int k = 0; k < 16; ++k)
        acc += x[node * 16 + k] * W0[k * 128 + f];
    out[idx] = fmaxf(acc, 0.0f);
}

// ---------------------------------------------------------------------------
// WMMA GEMM: C[n,128] (+)= A[n,128] @ Wt^T   (Wt stored [N=128,K=128] bf16)
// Compile-time ADD/BIAS/RELU -> branch-free epilogue (uniform tail guard).
// Block = 256 thr (8 waves), 16x128 C tile. A tile fetched by TDM (one wave
// issues tensor_load_to_lds; everyone converts fp32->bf16 in LDS after the
// barrier), then 4x v_wmma_f32_16x16x32_bf16 per wave.
// ---------------------------------------------------------------------------
template <int ADD, int BIAS, int RELU>
__global__ void __launch_bounds__(256)
k_gemm_wmma(const float* __restrict__ A, const unsigned short* __restrict__ Wt,
            float* __restrict__ C, const float* __restrict__ bias, int n_rows) {
    __shared__ __align__(16) unsigned short sA[16 * 128];   // bf16 A tile
    const int tid  = threadIdx.x;
    const int row0 = blockIdx.x * 16;

#ifdef HAVE_TDM
    __shared__ __align__(16) float sF[16 * 128];            // fp32 staging (TDM dest)
    if (tid < 32) {
        // rows past the tensor edge are zero-filled by the TDM OOB rule
        unsigned rows_left = (unsigned)(n_rows - row0);
        if (rows_left > 16u) rows_left = 16u;
        tdm_load_tile_f32(&A[(long long)row0 * 128],
                          (unsigned)(uintptr_t)(void*)sF,
                          /*tensor_d0=*/128, /*tensor_d1=*/rows_left,
                          /*stride_d0=*/128, /*tile_d0=*/128, /*tile_d1=*/16);
        tdm_wait();
    }
    __syncthreads();
    {
        int base = tid * 8;                                 // 256 thr * 8 = 2048
#pragma unroll
        for (int i = 0; i < 8; ++i) sA[base + i] = f2bf(sF[base + i]);
    }
#else
    {
        int base = tid * 8;
        int r = base >> 7;
        int k = base & 127;
        int row = row0 + r;
        if (row >= n_rows) row = n_rows - 1;
        const float* srcp = &A[(long long)row * 128 + k];
#pragma unroll
        for (int i = 0; i < 8; ++i) sA[base + i] = f2bf(srcp[i]);
    }
#endif
    __syncthreads();

    const int wave = tid >> 5;       // 0..7 -> which 16-col N tile
    const int lane = tid & 31;
    const int m    = lane & 15;      // A row / C col within tile
    const int hi   = lane >> 4;      // lane-half selector
    const int n    = wave * 16 + m;  // global output column

    v8f acc = {};
    Frag16 a, b;
#pragma unroll
    for (int kb = 0; kb < 128; kb += 32) {
        // A fragment: halves 0..7 -> K = kb + hi*8 + 0..7 ; halves 8..15 -> +16
        const v4u* pa = (const v4u*)&sA[m * 128 + kb + hi * 8];
        a.q[0] = pa[0];
        a.q[1] = pa[2];
        // B fragment: halves 0..7 -> K = kb + hi*8 + 0..7 ; halves 8..15 -> +16
        const v4u* pb = (const v4u*)&Wt[n * 128 + kb + hi * 8];
        b.q[0] = pb[0];
        b.q[1] = pb[2];
        acc = __builtin_amdgcn_wmma_f32_16x16x32_bf16(
            /*neg_a=*/false, a.v, /*neg_b=*/false, b.v,
            /*c_mod=*/(short)0, acc, /*reuse_a=*/false, /*reuse_b=*/false);
    }

    // C/D layout: acc[r] -> row = row0 + r + hi*8, col = n
    const float bv = BIAS ? bias[n] : 0.0f;
    float* cp = C + (long long)(row0 + hi * 8) * 128 + n;
    if (row0 + 16 <= n_rows) {
        // full tile: uniform branch, straight-line clause-able stores
#pragma unroll
        for (int r = 0; r < 8; ++r) {
            float v = acc[r];
            if (ADD)  v += cp[r * 128];
            if (BIAS) v += bv;
            if (RELU) v = fmaxf(v, 0.0f);
            cp[r * 128] = v;
        }
    } else {
        // tail tile only
#pragma unroll
        for (int r = 0; r < 8; ++r) {
            int row = row0 + r + hi * 8;
            if (row < n_rows) {
                float v = acc[r];
                if (ADD)  v += cp[r * 128];
                if (BIAS) v += bv;
                if (RELU) v = fmaxf(v, 0.0f);
                cp[r * 128] = v;
            }
        }
    }
}

// ---------------------------------------------------------------------------
// Propagation: t[dst] += h[src] * (dinv[src]*dinv[dst]); one wave per edge,
// each lane handles 4 features (float4 gather + 4 float atomics).
// ---------------------------------------------------------------------------
__global__ void k_prop(const int* __restrict__ src, const int* __restrict__ dst,
                       const float* __restrict__ dinv, const float* __restrict__ h,
                       float* __restrict__ t, int n_edges) {
    long long idx = (long long)blockIdx.x * blockDim.x + threadIdx.x;
    if (idx >= (long long)n_edges * 32) return;
    int e = (int)(idx >> 5);
    int g = ((int)idx & 31) * 4;
    int s = src[e], d = dst[e];
    float w = dinv[s] * dinv[d];
    const float4 hv = *(const float4*)&h[(long long)s * 128 + g];
    float* outp = &t[(long long)d * 128 + g];
    atomicAdd(outp + 0, hv.x * w);
    atomicAdd(outp + 1, hv.y * w);
    atomicAdd(outp + 2, hv.z * w);
    atomicAdd(outp + 3, hv.w * w);
}

// ---------------------------------------------------------------------------
// lin3: [N,128] @ [128,1] + b3, ReLU -> [N,1]
// ---------------------------------------------------------------------------
__global__ void k_lin3(const float* __restrict__ h, const float* __restrict__ W3,
                       const float* __restrict__ b3, float* __restrict__ out, int n_nodes) {
    int nid = blockIdx.x * blockDim.x + threadIdx.x;
    if (nid >= n_nodes) return;
    const float4* hp = (const float4*)&h[(long long)nid * 128];
    const float4* wp = (const float4*)W3;
    float a = 0.0f;
#pragma unroll 8
    for (int k = 0; k < 32; ++k) {
        float4 hv = hp[k], wv = wp[k];
        a += hv.x * wv.x + hv.y * wv.y + hv.z * wv.z + hv.w * wv.w;
    }
    out[nid] = fmaxf(a + b3[0], 0.0f);
}

// ---------------------------------------------------------------------------
// Host orchestration
// ---------------------------------------------------------------------------
extern "C" void kernel_launch(void* const* d_in, const int* in_sizes, int n_in,
                              void* d_out, int out_size, void* d_ws, size_t ws_size,
                              hipStream_t stream) {
    const float* x   = (const float*)d_in[0];
    const int*   ei  = (const int*)  d_in[1];
    const float* W0  = (const float*)d_in[2];
    const float* b0  = (const float*)d_in[3];
    const float* c1W = (const float*)d_in[4];
    const float* c1b = (const float*)d_in[5];
    const float* c2W = (const float*)d_in[6];
    const float* c2b = (const float*)d_in[7];
    const float* W1  = (const float*)d_in[8];
    const float* b1  = (const float*)d_in[9];
    const float* W2  = (const float*)d_in[10];
    const float* b2  = (const float*)d_in[11];
    const float* W3  = (const float*)d_in[12];
    const float* b3  = (const float*)d_in[13];
    float* out = (float*)d_out;

    const int n = in_sizes[0] / 16;   // 100000 nodes
    const int e = in_sizes[1] / 2;    // 600000 edges
    const int* srcI = ei;             // edge_index row 0
    const int* dstI = ei + e;         // edge_index row 1

    const long long NF = (long long)n * 128;

    // workspace layout
    float* bufA = (float*)d_ws;
    float* bufB = bufA + NF;
    float* bufT = bufB + NF;
    float* deg  = bufT + NF;
    float* dinv = deg + n;
    unsigned short* wt = (unsigned short*)(dinv + n);   // 10 matrices of 128x128 bf16
    const int WMAT = 128 * 128;

    const int ZB = 4096;
    const int gemmBlocks = (n + 15) / 16;
    const int propBlocks = (int)(((long long)e * 32 + 255) / 256);

    // --- weight conversion (fp32 [K,N] -> bf16 [N,K]) ---
    for (int m = 0; m < 4; ++m)
        k_convert_wt<<<WMAT / 256, 256, 0, stream>>>(c1W + (long long)m * WMAT, wt + (long long)m * WMAT, 128, 128);
    for (int m = 0; m < 4; ++m)
        k_convert_wt<<<WMAT / 256, 256, 0, stream>>>(c2W + (long long)m * WMAT, wt + (long long)(4 + m) * WMAT, 128, 128);
    k_convert_wt<<<WMAT / 256, 256, 0, stream>>>(W1, wt + (long long)8 * WMAT, 128, 128);
    k_convert_wt<<<WMAT / 256, 256, 0, stream>>>(W2, wt + (long long)9 * WMAT, 128, 128);

    // --- degrees & symmetric norm ---
    k_zero<<<ZB, 256, 0, stream>>>(deg, n);
    k_deg<<<(e + 255) / 256, 256, 0, stream>>>(dstI, deg, e);
    k_dinv<<<(n + 255) / 256, 256, 0, stream>>>(deg, dinv, n);

    // --- lin0 + ReLU -> bufA ---
    k_lin0<<<(int)((NF + 255) / 256), 256, 0, stream>>>(x, W0, b0, bufA, n);

    // gemm dispatch over the 4 used epilogue variants
    auto gemm = [&](const float* Ain, const unsigned short* Wm, float* Cout,
                    const float* bias, int add, int relu) {
        if (add == 0 && bias == nullptr)
            k_gemm_wmma<0, 0, 0><<<gemmBlocks, 256, 0, stream>>>(Ain, Wm, Cout, nullptr, n);
        else if (add == 1 && bias == nullptr)
            k_gemm_wmma<1, 0, 0><<<gemmBlocks, 256, 0, stream>>>(Ain, Wm, Cout, nullptr, n);
        else if (add == 1)
            k_gemm_wmma<1, 1, 1><<<gemmBlocks, 256, 0, stream>>>(Ain, Wm, Cout, bias, n);
        else
            k_gemm_wmma<0, 1, 1><<<gemmBlocks, 256, 0, stream>>>(Ain, Wm, Cout, bias, n);
        (void)relu;
    };
    auto zero = [&](float* p) { k_zero<<<ZB, 256, 0, stream>>>(p, NF); };
    auto prop = [&](const float* h, float* t) {
        k_prop<<<propBlocks, 256, 0, stream>>>(srcI, dstI, dinv, h, t, e);
    };

    // --- TAGConv layer 1: in=bufA, out=bufB, temp=bufT ---
    gemm(bufA, wt + 0 * WMAT, bufB, nullptr, 0, 0);
    zero(bufT); prop(bufA, bufT);
    gemm(bufT, wt + 1 * WMAT, bufB, nullptr, 1, 0);
    zero(bufA); prop(bufT, bufA);
    gemm(bufA, wt + 2 * WMAT, bufB, nullptr, 1, 0);
    zero(bufT); prop(bufA, bufT);
    gemm(bufT, wt + 3 * WMAT, bufB, c1b, 1, 1);        // +bias, ReLU -> bufB

    // --- TAGConv layer 2: in=bufB, out=bufA ---
    gemm(bufB, wt + 4 * WMAT, bufA, nullptr, 0, 0);
    zero(bufT); prop(bufB, bufT);
    gemm(bufT, wt + 5 * WMAT, bufA, nullptr, 1, 0);
    zero(bufB); prop(bufT, bufB);
    gemm(bufB, wt + 6 * WMAT, bufA, nullptr, 1, 0);
    zero(bufT); prop(bufB, bufT);
    gemm(bufT, wt + 7 * WMAT, bufA, c2b, 1, 1);        // +bias, ReLU -> bufA

    // --- lin1, lin2 (WMMA), lin3 ---
    gemm(bufA, wt + 8 * WMAT, bufB, b1, 0, 1);         // -> bufB
    gemm(bufB, wt + 9 * WMAT, bufA, b2, 0, 1);         // -> bufA
    k_lin3<<<(n + 255) / 256, 256, 0, stream>>>(bufA, W3, b3, out, n);
}
